// Conan_34720515621618
// MI455X (gfx1250) — matrix operand order repaired and checked by reading, not compiled, not run on hardware
//
#include <hip/hip_runtime.h>
#include <hip/hip_bf16.h>
#include <math.h>

typedef __attribute__((ext_vector_type(16))) __bf16 v16bf;
typedef __attribute__((ext_vector_type(8)))  float  v8f;
typedef __attribute__((ext_vector_type(4)))  float  f32x4;

#define NPTS   20000
#define BATCH  4
#define PTOT   (NPTS*BATCH)
#define DIN    2048
#define KSEL   10
#define GROUPS (PTOT/64)   /* 1250 groups of 64 points */
#define WPB    4           /* waves per block */
#define BLK1   (WPB*32)

__device__ __forceinline__ float bn_inv_scale() { return 1.0f / sqrtf(1.0f + 1e-5f); }

__device__ __forceinline__ v8f wmma_bf16(v16bf a, v16bf b, v8f c) {
  return __builtin_amdgcn_wmma_f32_16x16x32_bf16(false, a, false, b, (short)0, c, false, false);
}

// ---------------- Kernel 0: split W1 (32x2048 fp32) into bf16 hi/lo ----------------
__global__ void k_split_w1(const float* __restrict__ W1,
                           __bf16* __restrict__ hi, __bf16* __restrict__ lo) {
  int i = blockIdx.x * blockDim.x + threadIdx.x;
  if (i < 32 * DIN) {
    float f = W1[i];
    __bf16 h = (__bf16)f;
    hi[i] = h;
    lo[i] = (__bf16)(f - (float)h);
  }
}

// ---------------- Kernel 1: fused per-point backbone (WMMA GEMM + tiny MLP) --------
__global__ __launch_bounds__(BLK1) void k_backbone(
    const float* __restrict__ x,
    const __bf16* __restrict__ w1hi, const __bf16* __restrict__ w1lo,
    const float* __restrict__ b1, const float* __restrict__ g1, const float* __restrict__ be1,
    const float* __restrict__ W2, const float* __restrict__ b2, const float* __restrict__ g2, const float* __restrict__ be2,
    const float* __restrict__ W3, const float* __restrict__ b3, const float* __restrict__ g3, const float* __restrict__ be3,
    const float* __restrict__ Ws, const float* __restrict__ bs, const float* __restrict__ gs, const float* __restrict__ bes,
    float* __restrict__ h3g, float* __restrict__ sout)
{
  __shared__ float h1s[WPB][64][33];   // padded stride 33: conflict-free column reads
  __shared__ float cW2[8][32];
  __shared__ float cW3[32][8];
  __shared__ float cWs[32];
  __shared__ float sc1[32], sh1[32], sc3[32], sh3[32];
  __shared__ float sc2[8],  sh2[8];
  __shared__ float shead[2];

  const int tid = threadIdx.x;
  const int wv  = tid >> 5;
  const int l   = tid & 31;
  const int ln  = l & 15;
  const int hl  = l >> 4;
  const float invs = bn_inv_scale();

  if (tid < 32) {
    float s1 = g1[tid] * invs; sc1[tid] = s1; sh1[tid] = s1 * b1[tid] + be1[tid];
    float s3 = g3[tid] * invs; sc3[tid] = s3; sh3[tid] = s3 * b3[tid] + be3[tid];
    cWs[tid] = Ws[tid];
    for (int k = 0; k < 8; ++k) cW3[tid][k] = W3[tid * 8 + k];
  }
  if (tid < 8) {
    float s2 = g2[tid] * invs; sc2[tid] = s2; sh2[tid] = s2 * b2[tid] + be2[tid];
    for (int k = 0; k < 32; ++k) cW2[tid][k] = W2[tid * 32 + k];
  }
  if (tid == 0) { float ssc = gs[0] * invs; shead[0] = ssc; shead[1] = ssc * bs[0] + bes[0]; }
  __syncthreads();

  const v8f vzero = {0.f,0.f,0.f,0.f,0.f,0.f,0.f,0.f};

  for (int g = blockIdx.x * WPB + wv; g < GROUPS; g += gridDim.x * WPB) {
    v8f acc[4][2];
    #pragma unroll
    for (int t = 0; t < 4; ++t) { acc[t][0] = vzero; acc[t][1] = vzero; }

    const long p0 = (long)g * 64;

    #pragma unroll 1
    for (int kb = 0; kb < DIN / 32; ++kb) {
      // B fragments: lane = output channel (n = ln [+16]), 16 contiguous bf16 per lane.
      // W1 bf16 copies are small & hot: default (RT) caching so they live in WGP$/L2.
      const int koff = kb * 32 + hl * 16;
      v16bf bh0 = *(const v16bf*)(w1hi + (size_t)ln        * DIN + koff);
      v16bf bl0 = *(const v16bf*)(w1lo + (size_t)ln        * DIN + koff);
      v16bf bh1 = *(const v16bf*)(w1hi + (size_t)(16 + ln) * DIN + koff);
      v16bf bl1 = *(const v16bf*)(w1lo + (size_t)(16 + ln) * DIN + koff);

      #pragma unroll
      for (int t = 0; t < 4; ++t) {
        // A fragment: lane row m = ln, K elements {k0..k0+15}, k0 = hl*16.
        // x (655 MB) is streamed exactly once -> non-temporal so it doesn't evict the
        // resident W1/bf16 + h3/s working set from L2.
        const float* xr = x + (size_t)(p0 + t * 16 + ln) * DIN + kb * 32 + hl * 16;
        f32x4 a0 = __builtin_nontemporal_load((const f32x4*)(xr));
        f32x4 a1 = __builtin_nontemporal_load((const f32x4*)(xr + 4));
        f32x4 a2 = __builtin_nontemporal_load((const f32x4*)(xr + 8));
        f32x4 a3 = __builtin_nontemporal_load((const f32x4*)(xr + 12));
        v16bf ah, al;
        #pragma unroll
        for (int j = 0; j < 4; ++j) {
          float f0 = a0[j]; __bf16 h0 = (__bf16)f0; ah[j]      = h0; al[j]      = (__bf16)(f0 - (float)h0);
          float f1 = a1[j]; __bf16 h1 = (__bf16)f1; ah[4 + j]  = h1; al[4 + j]  = (__bf16)(f1 - (float)h1);
          float f2 = a2[j]; __bf16 h2 = (__bf16)f2; ah[8 + j]  = h2; al[8 + j]  = (__bf16)(f2 - (float)h2);
          float f3 = a3[j]; __bf16 h3 = (__bf16)f3; ah[12 + j] = h3; al[12 + j] = (__bf16)(f3 - (float)h3);
        }
        // split-precision fp32 emulation: hi*Whi + hi*Wlo + lo*Whi (f32 accumulate)
        acc[t][0] = wmma_bf16(ah, bh0, acc[t][0]);
        acc[t][0] = wmma_bf16(ah, bl0, acc[t][0]);
        acc[t][0] = wmma_bf16(al, bh0, acc[t][0]);
        acc[t][1] = wmma_bf16(ah, bh1, acc[t][1]);
        acc[t][1] = wmma_bf16(ah, bl1, acc[t][1]);
        acc[t][1] = wmma_bf16(al, bh1, acc[t][1]);
      }
    }

    // BN + ReLU (layer 1) -> LDS. C layout: N=ln (channel), M = hl*8 + r (point row).
    #pragma unroll
    for (int t = 0; t < 4; ++t) {
      #pragma unroll
      for (int u = 0; u < 2; ++u) {
        const int c = u * 16 + ln;
        const float sc = sc1[c], sh = sh1[c];
        #pragma unroll
        for (int r = 0; r < 8; ++r) {
          const int pl = t * 16 + hl * 8 + r;
          h1s[wv][pl][c] = fmaxf(0.f, sc * acc[t][u][r] + sh);
        }
      }
    }
    // wave-internal cross-lane LDS handoff (no block barrier: per-wave trip counts differ)
    asm volatile("s_wait_dscnt 0" ::: "memory");

    const float ssc = shead[0], ssh = shead[1];
    #pragma unroll 1
    for (int pp = 0; pp < 2; ++pp) {
      const int  pl = l * 2 + pp;
      const long p  = p0 + pl;
      float h2[8];
      #pragma unroll
      for (int c2 = 0; c2 < 8; ++c2) {
        float a = 0.f;
        #pragma unroll
        for (int k = 0; k < 32; ++k) a += h1s[wv][pl][k] * cW2[c2][k];
        h2[c2] = fmaxf(0.f, sc2[c2] * a + sh2[c2]);
      }
      float sacc = 0.f;
      #pragma unroll 1
      for (int c3 = 0; c3 < 32; ++c3) {
        float a = 0.f;
        #pragma unroll
        for (int k = 0; k < 8; ++k) a += h2[k] * cW3[c3][k];
        float hv = fmaxf(0.f, sc3[c3] * a + sh3[c3]);
        h3g[p * 32 + c3] = hv;
        sacc += cWs[c3] * hv;
      }
      sout[p] = fmaxf(0.f, ssc * sacc + ssh);
    }
    asm volatile("s_wait_dscnt 0" ::: "memory");
  }
}

// ---------------- Kernel 2: stable-argsort bottom-K / top-K per batch --------------
__device__ __forceinline__ bool pless(float v1, int i1, float v2, int i2) {
  return (v1 < v2) || (v1 == v2 && i1 < i2);   // ascending stable-sort comparator
}

__global__ __launch_bounds__(32) void k_select(const float* __restrict__ s,
                                               int* __restrict__ sel, float* __restrict__ sg) {
  const int b = blockIdx.x;
  const int l = threadIdx.x;
  const float* sb = s + (size_t)b * NPTS;

  float bv[KSEL], tv[KSEL]; int bi[KSEL], ti[KSEL];
  for (int k = 0; k < KSEL; ++k) { bv[k] = 3.4e38f; bi[k] = 0x7fffffff; tv[k] = -3.4e38f; ti[k] = -1; }

  for (int i = l; i < NPTS; i += 32) {
    float v = sb[i];
    if (pless(v, i, bv[KSEL - 1], bi[KSEL - 1])) {           // 10 smallest, ascending
      int k = KSEL - 1;
      while (k > 0 && pless(v, i, bv[k - 1], bi[k - 1])) { bv[k] = bv[k - 1]; bi[k] = bi[k - 1]; --k; }
      bv[k] = v; bi[k] = i;
    }
    if (pless(tv[0], ti[0], v, i)) {                         // 10 largest, stored ascending
      int k = 0;
      while (k < KSEL - 1 && pless(tv[k + 1], ti[k + 1], v, i)) { tv[k] = tv[k + 1]; ti[k] = ti[k + 1]; ++k; }
      tv[k] = v; ti[k] = i;
    }
  }

  __shared__ float lbv[32][KSEL], ltv[32][KSEL];
  __shared__ int   lbi[32][KSEL], lti[32][KSEL];
  for (int k = 0; k < KSEL; ++k) { lbv[l][k] = bv[k]; lbi[l][k] = bi[k]; ltv[l][k] = tv[k]; lti[l][k] = ti[k]; }
  __syncthreads();

  if (l == 0) {
    float fbv[KSEL], ftv[KSEL]; int fbi[KSEL], fti[KSEL];
    for (int k = 0; k < KSEL; ++k) { fbv[k] = 3.4e38f; fbi[k] = 0x7fffffff; ftv[k] = -3.4e38f; fti[k] = -1; }
    for (int lane = 0; lane < 32; ++lane) {
      for (int k = 0; k < KSEL; ++k) {
        float v = lbv[lane][k]; int i = lbi[lane][k];
        if (pless(v, i, fbv[KSEL - 1], fbi[KSEL - 1])) {
          int m = KSEL - 1;
          while (m > 0 && pless(v, i, fbv[m - 1], fbi[m - 1])) { fbv[m] = fbv[m - 1]; fbi[m] = fbi[m - 1]; --m; }
          fbv[m] = v; fbi[m] = i;
        }
        v = ltv[lane][k]; i = lti[lane][k];
        if (pless(ftv[0], fti[0], v, i)) {
          int m = 0;
          while (m < KSEL - 1 && pless(ftv[m + 1], fti[m + 1], v, i)) { ftv[m] = ftv[m + 1]; fti[m] = fti[m + 1]; ++m; }
          ftv[m] = v; fti[m] = i;
        }
      }
    }
    for (int j = 0; j < KSEL; ++j) {
      sel[b * 2 * KSEL + j]        = b * NPTS + fbi[j];  sg[b * 2 * KSEL + j]        = fbv[j];
      sel[b * 2 * KSEL + KSEL + j] = b * NPTS + fti[j];  sg[b * 2 * KSEL + KSEL + j] = ftv[j];
    }
  }
}

// ---------------- Kernel 3: gather + classifier head + sigmoid ---------------------
__global__ __launch_bounds__(32) void k_head(
    const float* __restrict__ h3g, const int* __restrict__ sel, const float* __restrict__ sgv,
    const float* __restrict__ Wf1, const float* __restrict__ bf1, const float* __restrict__ gf1, const float* __restrict__ bef1,
    const float* __restrict__ Wf2, const float* __restrict__ bf2, const float* __restrict__ gf2, const float* __restrict__ bef2,
    const float* __restrict__ Wf3, const float* __restrict__ bf3, float* __restrict__ out)
{
  const int b = blockIdx.x;
  const int l = threadIdx.x;              // l = channel 0..31
  __shared__ float feat[692];
  __shared__ float z1[32], z2[32];
  const float invs = bn_inv_scale();

  float acc = 0.f;
  for (int j = 0; j < 2 * KSEL; ++j) {
    float v = h3g[(size_t)sel[b * 2 * KSEL + j] * 32 + l];
    feat[52 + l * 20 + j] = v;            // out.reshape(B,-1): index c*20+j
    acc += v;
  }
  feat[20 + l] = acc * (1.f / 20.f);      // avg
  if (l < 20) feat[l] = sgv[b * 2 * KSEL + l];   // sg
  __syncthreads();

  float a1 = 0.f;
  for (int k = 0; k < 692; ++k) a1 += feat[k] * Wf1[l * 692 + k];
  z1[l] = fmaxf(0.f, gf1[l] * invs * (a1 + bf1[l]) + bef1[l]);
  __syncthreads();

  float a2 = 0.f;
  for (int k = 0; k < 32; ++k) a2 += z1[k] * Wf2[l * 32 + k];
  z2[l] = fmaxf(0.f, gf2[l] * invs * (a2 + bf2[l]) + bef2[l]);
  __syncthreads();

  if (l == 0) {
    float lg = 0.f;
    for (int k = 0; k < 32; ++k) lg += z2[k] * Wf3[k];
    lg += bf3[0];
    out[b] = 1.f / (1.f + expf(-lg));
  }
}

// ---------------- Host launch ------------------------------------------------------
extern "C" void kernel_launch(void* const* d_in, const int* in_sizes, int n_in,
                              void* d_out, int out_size, void* d_ws, size_t ws_size,
                              hipStream_t stream) {
  const float* x   = (const float*)d_in[0];
  const float* W1  = (const float*)d_in[1];
  const float* b1  = (const float*)d_in[2];
  const float* g1  = (const float*)d_in[3];
  const float* be1 = (const float*)d_in[4];
  const float* W2  = (const float*)d_in[5];
  const float* b2  = (const float*)d_in[6];
  const float* g2  = (const float*)d_in[7];
  const float* be2 = (const float*)d_in[8];
  const float* W3  = (const float*)d_in[9];
  const float* b3  = (const float*)d_in[10];
  const float* g3  = (const float*)d_in[11];
  const float* be3 = (const float*)d_in[12];
  const float* Ws  = (const float*)d_in[13];
  const float* bs  = (const float*)d_in[14];
  const float* gs  = (const float*)d_in[15];
  const float* bes = (const float*)d_in[16];
  const float* Wf1 = (const float*)d_in[17];
  const float* bf1 = (const float*)d_in[18];
  const float* gf1 = (const float*)d_in[19];
  const float* bef1= (const float*)d_in[20];
  const float* Wf2 = (const float*)d_in[21];
  const float* bf2 = (const float*)d_in[22];
  const float* gf2 = (const float*)d_in[23];
  const float* bef2= (const float*)d_in[24];
  const float* Wf3 = (const float*)d_in[25];
  const float* bf3 = (const float*)d_in[26];

  char* ws = (char*)d_ws;
  const size_t OFF_W1LO = (size_t)32 * DIN * 2;                 // 131072
  const size_t OFF_H3G  = OFF_W1LO + (size_t)32 * DIN * 2;      // 262144
  const size_t OFF_S    = OFF_H3G + (size_t)PTOT * 32 * 4;      // + 10.24 MB
  const size_t OFF_SEL  = OFF_S   + (size_t)PTOT * 4;
  const size_t OFF_SG   = OFF_SEL + (size_t)BATCH * 2 * KSEL * 4;

  __bf16* w1hi = (__bf16*)ws;
  __bf16* w1lo = (__bf16*)(ws + OFF_W1LO);
  float*  h3g  = (float*)(ws + OFF_H3G);
  float*  sout = (float*)(ws + OFF_S);
  int*    sel  = (int*)  (ws + OFF_SEL);
  float*  sgv  = (float*)(ws + OFF_SG);

  k_split_w1<<<(32 * DIN + 255) / 256, 256, 0, stream>>>(W1, w1hi, w1lo);

  const int grid1 = (GROUPS + WPB - 1) / WPB;   // 313 blocks x 4 waves = 1252 wave slots
  k_backbone<<<grid1, BLK1, 0, stream>>>(x, w1hi, w1lo,
      b1, g1, be1, W2, b2, g2, be2, W3, b3, g3, be3, Ws, bs, gs, bes,
      h3g, sout);

  k_select<<<BATCH, 32, 0, stream>>>(sout, sel, sgv);

  k_head<<<BATCH, 32, 0, stream>>>(h3g, sel, sgv,
      Wf1, bf1, gf1, bef1, Wf2, bf2, gf2, bef2, Wf3, bf3, (float*)d_out);
}
